// _EGLayer_83416854823130
// MI455X (gfx1250) — matrix operand order repaired and checked by reading, not compiled, not run on hardware
//
#include <hip/hip_runtime.h>
#include <math.h>

// ---------------------------------------------------------------------------
// EGNN layer for MI455X (gfx1250, wave32). All five GEMMs run on
// v_wmma_f32_16x16x32_f16; 16 edges (or nodes) per wave; messages live only
// in accumulator VGPRs; scatter-add via hw fp32 global atomics (L2-resident).
// ---------------------------------------------------------------------------

typedef _Float16 v16h __attribute__((ext_vector_type(16)));
typedef float    v8f  __attribute__((ext_vector_type(8)));

union AF { v16h v; unsigned u[8]; };
union PK { unsigned u; _Float16 h[2]; };

#define WAVES 4          // waves (tiles) per block, 128 threads
#define D     64
#define GD    64

__device__ __forceinline__ void lds_fence() {
  asm volatile("s_wait_dscnt 0" ::: "memory");
}

__device__ __forceinline__ float siluf(float v) { return v / (1.0f + __expf(-v)); }

__device__ __forceinline__ v8f splat8(float b) {
  v8f v;
#pragma unroll
  for (int i = 0; i < 8; ++i) v[i] = b;
  return v;
}

// A fragment (16x32 f16, M=lane&15, K split by half) from an LDS row.
__device__ __forceinline__ v16h a_frag_lds(const _Float16* row, int k0, int half) {
  AF a;
#pragma unroll
  for (int j = 0; j < 8; ++j) {
    int K = k0 + 8 * half + ((j < 4) ? (2 * j) : (16 + 2 * (j - 4)));
    a.u[j] = *(const unsigned*)(row + K);   // f16 pair (K, K+1), 4B aligned
  }
  return a.v;
}

// B fragment (32x16 f16) from transposed f16 weights WT[N][K] (K contiguous).
__device__ __forceinline__ v16h b_frag(const _Float16* __restrict__ WT, int ldk,
                                       int nglob, int k0, int half) {
  AF b;
  const _Float16* p = WT + nglob * ldk + k0 + 8 * half;
#pragma unroll
  for (int j = 0; j < 8; ++j) b.u[j] = *(const unsigned*)(p + ((j < 4) ? (2 * j) : (16 + 2 * (j - 4))));
  return b.v;
}

__device__ __forceinline__ void atomic_add_f32(float* p, float v) {
  (void)__hip_atomic_fetch_add(p, v, __ATOMIC_RELAXED, __HIP_MEMORY_SCOPE_AGENT);
}

// ---------------------------------------------------------------------------
__global__ void zero_kernel(float* p, long n) {
  long i = (long)blockIdx.x * blockDim.x + threadIdx.x;
  long s = (long)gridDim.x * blockDim.x;
  for (; i < n; i += s) p[i] = 0.0f;
}

// Transpose weights to f16 [N][K] layout for B-fragment loads.
__global__ void prep_kernel(const float* __restrict__ Wm1, const float* __restrict__ Wm2,
                            const float* __restrict__ Wx1, const float* __restrict__ Wh1,
                            const float* __restrict__ Wh2, _Float16* __restrict__ wt) {
  int tid = blockIdx.x * blockDim.x + threadIdx.x;
  int stride = gridDim.x * blockDim.x;
  _Float16* m1 = wt;                 // 64 x 192
  _Float16* m2 = m1 + 64 * 192;      // 64 x 64
  _Float16* x1 = m2 + 64 * 64;       // 64 x 64
  _Float16* h1 = x1 + 64 * 64;       // 64 x 128
  _Float16* h2 = h1 + 64 * 128;      // 64 x 64
  for (int i = tid; i < 64 * 192; i += stride) { int n = i / 192, k = i % 192; m1[i] = (_Float16)Wm1[k * 64 + n]; }
  for (int i = tid; i < 64 * 64;  i += stride) { int n = i / 64,  k = i % 64;  m2[i] = (_Float16)Wm2[k * 64 + n]; }
  for (int i = tid; i < 64 * 64;  i += stride) { int n = i / 64,  k = i % 64;  x1[i] = (_Float16)Wx1[k * 64 + n]; }
  for (int i = tid; i < 64 * 128; i += stride) { int n = i / 128, k = i % 128; h1[i] = (_Float16)Wh1[k * 64 + n]; }
  for (int i = tid; i < 64 * 64;  i += stride) { int n = i / 64,  k = i % 64;  h2[i] = (_Float16)Wh2[k * 64 + n]; }
}

// ---------------------------------------------------------------------------
// Edge kernel: one 16-edge tile per wave32.
__global__ void __launch_bounds__(128) edge_kernel(
    const float* __restrict__ h, const float* __restrict__ x,
    const int* __restrict__ edst, const int* __restrict__ esrc,
    const float* __restrict__ means, const float* __restrict__ stds,
    const _Float16* __restrict__ Wm1T, const float* __restrict__ bm1,
    const _Float16* __restrict__ Wm2T, const float* __restrict__ bm2,
    const float* __restrict__ Wa, const float* __restrict__ ba,
    const _Float16* __restrict__ Wx1T, const float* __restrict__ bx1,
    const float* __restrict__ Wx2,
    float* __restrict__ node_msg, float* __restrict__ x_acc, int E) {
  __shared__ _Float16 s_feat[WAVES][16][192] __attribute__((aligned(16)));
  __shared__ _Float16 s_act[WAVES][16][64] __attribute__((aligned(16)));
  __shared__ int   s_nd[WAVES][16];
  __shared__ int   s_ns[WAVES][16];
  __shared__ float s_dist[WAVES][16];
  __shared__ float s_mask[WAVES][16];
  __shared__ float s_dir[WAVES][16][3];

  const int lane = threadIdx.x & 31;
  const int w    = threadIdx.x >> 5;
  const long e0  = ((long)blockIdx.x * WAVES + w) * 16;
  if (e0 >= E) return;

  const int nn   = lane & 15;
  const int half = lane >> 4;

  // --- edge geometry (one edge per lane 0..15) ---
  if (lane < 16) {
    long e = e0 + lane;
    bool valid = (e < E);
    if (!valid) e = 0;
    int nd = edst[e], ns = esrc[e];
    float dx = x[nd * 3 + 0] - x[ns * 3 + 0];
    float dy = x[nd * 3 + 1] - x[ns * 3 + 1];
    float dz = x[nd * 3 + 2] - x[ns * 3 + 2];
    float d = sqrtf(dx * dx + dy * dy + dz * dz);
    float inv = 1.0f / d;
    s_nd[w][lane] = nd; s_ns[w][lane] = ns;
    s_dist[w][lane] = d;
    s_mask[w][lane] = (valid && d <= 3.0f) ? 1.0f : 0.0f;
    s_dir[w][lane][0] = dx * inv;
    s_dir[w][lane][1] = dy * inv;
    s_dir[w][lane][2] = dz * inv;
  }
  lds_fence();

  // --- stage [h_dst | h_src | rbf] as f16, coalesced float2 gathers ---
  const float mean0 = means[lane], std0 = stds[lane];
  const float mean1 = means[lane + 32], std1 = stds[lane + 32];
#pragma unroll 4
  for (int r = 0; r < 16; ++r) {
    int nd = s_nd[w][r], ns = s_ns[w][r];
    float2 hd = ((const float2*)(h + (long)nd * D))[lane];
    float2 hs = ((const float2*)(h + (long)ns * D))[lane];
    s_feat[w][r][2 * lane + 0]      = (_Float16)hd.x;
    s_feat[w][r][2 * lane + 1]      = (_Float16)hd.y;
    s_feat[w][r][64 + 2 * lane + 0] = (_Float16)hs.x;
    s_feat[w][r][64 + 2 * lane + 1] = (_Float16)hs.y;
    float d = s_dist[w][r];
    float t0 = (d - mean0) / std0;
    float t1 = (d - mean1) / std1;
    s_feat[w][r][128 + lane]      = (_Float16)__expf(-0.5f * t0 * t0);
    s_feat[w][r][128 + lane + 32] = (_Float16)__expf(-0.5f * t1 * t1);
  }
  lds_fence();

  const _Float16* arow = &s_feat[w][nn][0];

  // --- layer 1: [16x192] @ Wm1 + bm1, SiLU  (24 WMMAs) ---
  v8f acc[4];
#pragma unroll
  for (int t = 0; t < 4; ++t) acc[t] = splat8(bm1[nn + 16 * t]);
#pragma unroll
  for (int kk = 0; kk < 6; ++kk) {
    v16h a = a_frag_lds(arow, 32 * kk, half);
#pragma unroll
    for (int t = 0; t < 4; ++t) {
      v16h b = b_frag(Wm1T, 192, 16 * t + nn, 32 * kk, half);
      acc[t] = __builtin_amdgcn_wmma_f32_16x16x32_f16(false, a, false, b, (short)0, acc[t], false, false);
    }
  }
#pragma unroll
  for (int t = 0; t < 4; ++t)
#pragma unroll
    for (int r = 0; r < 8; ++r)
      s_act[w][r + 8 * half][nn + 16 * t] = (_Float16)siluf(acc[t][r]);
  lds_fence();

  // --- layer 2: [16x64] @ Wm2 + bm2, SiLU  (8 WMMAs) ---
  const _Float16* mrow = &s_act[w][nn][0];
#pragma unroll
  for (int t = 0; t < 4; ++t) acc[t] = splat8(bm2[nn + 16 * t]);
#pragma unroll
  for (int kk = 0; kk < 2; ++kk) {
    v16h a = a_frag_lds(mrow, 32 * kk, half);
#pragma unroll
    for (int t = 0; t < 4; ++t) {
      v16h b = b_frag(Wm2T, 64, 16 * t + nn, 32 * kk, half);
      acc[t] = __builtin_amdgcn_wmma_f32_16x16x32_f16(false, a, false, b, (short)0, acc[t], false, false);
    }
  }
#pragma unroll
  for (int t = 0; t < 4; ++t)
#pragma unroll
    for (int r = 0; r < 8; ++r)
      acc[t][r] = siluf(acc[t][r]);

  // --- attention: sigmoid(m . Wa + ba) per edge, butterfly over 16 lanes ---
  const float wa0 = Wa[nn], wa1 = Wa[nn + 16], wa2 = Wa[nn + 32], wa3 = Wa[nn + 48];
  const float bav = ba[0];
  float scl[8];
#pragma unroll
  for (int r = 0; r < 8; ++r) {
    float p = acc[0][r] * wa0 + acc[1][r] * wa1 + acc[2][r] * wa2 + acc[3][r] * wa3;
    p += __shfl_xor(p, 1);
    p += __shfl_xor(p, 2);
    p += __shfl_xor(p, 4);
    p += __shfl_xor(p, 8);
    float att = 1.0f / (1.0f + __expf(-(p + bav)));
    scl[r] = att * s_mask[w][r + 8 * half];
  }
#pragma unroll
  for (int t = 0; t < 4; ++t)
#pragma unroll
    for (int r = 0; r < 8; ++r)
      acc[t][r] *= scl[r];

  // --- scatter messages to node_msg (hw fp32 atomics, L2-resident) ---
#pragma unroll
  for (int r = 0; r < 8; ++r) {
    int node = s_nd[w][r + 8 * half];
    long base = (long)node * D + nn;
#pragma unroll
    for (int t = 0; t < 4; ++t)
      atomic_add_f32(&node_msg[base + 16 * t], acc[t][r]);
  }

  // --- Wx1 branch: silu(m @ Wx1 + bx1) . Wx2, tanh, scatter displacement ---
#pragma unroll
  for (int t = 0; t < 4; ++t)
#pragma unroll
    for (int r = 0; r < 8; ++r)
      s_act[w][r + 8 * half][nn + 16 * t] = (_Float16)acc[t][r];
  lds_fence();

  v8f xac[4];
#pragma unroll
  for (int t = 0; t < 4; ++t) xac[t] = splat8(bx1[nn + 16 * t]);
#pragma unroll
  for (int kk = 0; kk < 2; ++kk) {
    v16h a = a_frag_lds(mrow, 32 * kk, half);
#pragma unroll
    for (int t = 0; t < 4; ++t) {
      v16h b = b_frag(Wx1T, 64, 16 * t + nn, 32 * kk, half);
      xac[t] = __builtin_amdgcn_wmma_f32_16x16x32_f16(false, a, false, b, (short)0, xac[t], false, false);
    }
  }
  const float wx0 = Wx2[nn], wx1 = Wx2[nn + 16], wx2 = Wx2[nn + 32], wx3 = Wx2[nn + 48];
#pragma unroll
  for (int r = 0; r < 8; ++r) {
    float p = siluf(xac[0][r]) * wx0 + siluf(xac[1][r]) * wx1 +
              siluf(xac[2][r]) * wx2 + siluf(xac[3][r]) * wx3;
    p += __shfl_xor(p, 1);
    p += __shfl_xor(p, 2);
    p += __shfl_xor(p, 4);
    p += __shfl_xor(p, 8);
    int M = r + 8 * half;
    float mag = tanhf(p) * s_mask[w][M];
    if (nn == 0) {
      int node = s_nd[w][M];
      atomic_add_f32(&x_acc[node * 3 + 0], s_dir[w][M][0] * mag);
      atomic_add_f32(&x_acc[node * 3 + 1], s_dir[w][M][1] * mag);
      atomic_add_f32(&x_acc[node * 3 + 2], s_dir[w][M][2] * mag);
    }
  }
}

// ---------------------------------------------------------------------------
// Node kernel: h_out = h + (silu([h|msg] @ Wh1 + bh1) @ Wh2 + bh2)
__global__ void __launch_bounds__(128) node_kernel(
    const float* __restrict__ h, const float* __restrict__ node_msg,
    const _Float16* __restrict__ Wh1T, const float* __restrict__ bh1,
    const _Float16* __restrict__ Wh2T, const float* __restrict__ bh2,
    float* __restrict__ hout, int N) {
  __shared__ _Float16 s_act[WAVES][16][64] __attribute__((aligned(16)));
  const int lane = threadIdx.x & 31;
  const int w = threadIdx.x >> 5;
  const int tile = blockIdx.x * WAVES + w;
  if ((long)tile * 16 >= N) return;
  const int nn = lane & 15, half = lane >> 4;
  int row = tile * 16 + nn;
  if (row >= N) row = N - 1;

  v8f acc[4];
#pragma unroll
  for (int t = 0; t < 4; ++t) acc[t] = splat8(bh1[nn + 16 * t]);
#pragma unroll
  for (int kk = 0; kk < 4; ++kk) {
    const float* src = (kk < 2) ? (h + (long)row * D + 32 * kk)
                                : (node_msg + (long)row * D + 32 * (kk - 2));
    AF a;
#pragma unroll
    for (int j = 0; j < 8; ++j) {
      int K = 8 * half + ((j < 4) ? (2 * j) : (16 + 2 * (j - 4)));
      float2 f = *(const float2*)(src + K);
      PK pk; pk.h[0] = (_Float16)f.x; pk.h[1] = (_Float16)f.y;
      a.u[j] = pk.u;
    }
#pragma unroll
    for (int t = 0; t < 4; ++t) {
      v16h b = b_frag(Wh1T, 128, 16 * t + nn, 32 * kk, half);
      acc[t] = __builtin_amdgcn_wmma_f32_16x16x32_f16(false, a.v, false, b, (short)0, acc[t], false, false);
    }
  }
#pragma unroll
  for (int t = 0; t < 4; ++t)
#pragma unroll
    for (int r = 0; r < 8; ++r)
      s_act[w][r + 8 * half][nn + 16 * t] = (_Float16)siluf(acc[t][r]);
  lds_fence();

  const _Float16* mrow = &s_act[w][nn][0];
#pragma unroll
  for (int t = 0; t < 4; ++t) acc[t] = splat8(bh2[nn + 16 * t]);
#pragma unroll
  for (int kk = 0; kk < 2; ++kk) {
    v16h a = a_frag_lds(mrow, 32 * kk, half);
#pragma unroll
    for (int t = 0; t < 4; ++t) {
      v16h b = b_frag(Wh2T, 64, 16 * t + nn, 32 * kk, half);
      acc[t] = __builtin_amdgcn_wmma_f32_16x16x32_f16(false, a, false, b, (short)0, acc[t], false, false);
    }
  }
#pragma unroll
  for (int t = 0; t < 4; ++t)
#pragma unroll
    for (int r = 0; r < 8; ++r) {
      int m = tile * 16 + r + 8 * half;
      if (m < N) {
        long idx = (long)m * D + nn + 16 * t;
        hout[idx] = h[idx] + acc[t][r];
      }
    }
}

__global__ void xout_kernel(const float* __restrict__ x, const float* __restrict__ x_acc,
                            float* __restrict__ xout, int n3) {
  int i = blockIdx.x * blockDim.x + threadIdx.x;
  if (i < n3) xout[i] = x[i] + x_acc[i];
}

// ---------------------------------------------------------------------------
extern "C" void kernel_launch(void* const* d_in, const int* in_sizes, int n_in,
                              void* d_out, int out_size, void* d_ws, size_t ws_size,
                              hipStream_t stream) {
  const float* h     = (const float*)d_in[0];
  const float* x     = (const float*)d_in[1];
  const int*   edst  = (const int*)d_in[2];
  const int*   esrc  = (const int*)d_in[3];
  const float* means = (const float*)d_in[4];
  const float* stds  = (const float*)d_in[5];
  const float* Wm1   = (const float*)d_in[6];
  const float* bm1   = (const float*)d_in[7];
  const float* Wm2   = (const float*)d_in[8];
  const float* bm2   = (const float*)d_in[9];
  const float* Wa    = (const float*)d_in[10];
  const float* ba    = (const float*)d_in[11];
  const float* Wx1   = (const float*)d_in[12];
  const float* bx1   = (const float*)d_in[13];
  const float* Wx2   = (const float*)d_in[14];
  const float* Wh1   = (const float*)d_in[15];
  const float* bh1   = (const float*)d_in[16];
  const float* Wh2   = (const float*)d_in[17];
  const float* bh2   = (const float*)d_in[18];

  const int N = in_sizes[0] / D;
  const int E = in_sizes[2];

  float* out_h = (float*)d_out;
  float* out_x = out_h + (size_t)N * D;

  // workspace: node_msg [N*64] f32 | x_acc [N*3] f32 | f16 transposed weights
  float* node_msg = (float*)d_ws;
  float* x_acc = node_msg + (size_t)N * D;
  _Float16* wt = (_Float16*)(x_acc + (size_t)N * 3);
  _Float16* Wm1T = wt;
  _Float16* Wm2T = Wm1T + 64 * 192;
  _Float16* Wx1T = Wm2T + 64 * 64;
  _Float16* Wh1T = Wx1T + 64 * 64;
  _Float16* Wh2T = Wh1T + 64 * 128;

  long nz = (long)N * D + (long)N * 3;
  zero_kernel<<<2048, 256, 0, stream>>>(node_msg, nz);
  prep_kernel<<<64, 256, 0, stream>>>(Wm1, Wm2, Wx1, Wh1, Wh2, wt);

  int etiles = (E + 15) / 16;
  int eblocks = (etiles + WAVES - 1) / WAVES;
  edge_kernel<<<eblocks, 128, 0, stream>>>(h, x, edst, esrc, means, stds,
                                           Wm1T, bm1, Wm2T, bm2, Wa, ba,
                                           Wx1T, bx1, Wx2, node_msg, x_acc, E);

  int ntiles = (N + 15) / 16;
  int nblocks = (ntiles + WAVES - 1) / WAVES;
  node_kernel<<<nblocks, 128, 0, stream>>>(h, node_msg, Wh1T, bh1, Wh2T, bh2, out_h, N);

  int n3 = N * 3;
  xout_kernel<<<(n3 + 255) / 256, 256, 0, stream>>>(x, x_acc, out_x, n3);
}
